// EntropySampler_10634339025304
// MI455X (gfx1250) — matrix operand order repaired and checked by reading, not compiled, not run on hardware
//
#include <hip/hip_runtime.h>

// ---------------- problem constants (from reference) ----------------
constexpr int kB = 8;        // batch
constexpr int kN = 4096;     // tokens
constexpr int kD = 256;      // dim
constexpr int kSampleK = 256;
constexpr int kRowsPerBlock = 128;           // 8 waves x 16 rows
constexpr int kNumTiles = kN / 16;           // column tiles
constexpr int kLdsStride = 264;              // 256 + 8 pad (bf16 elems) -> bank spread

typedef __attribute__((ext_vector_type(16))) __bf16 v16bf;
typedef __attribute__((ext_vector_type(8)))  float  v8f;
typedef __attribute__((ext_vector_type(4)))  int    v4i;

union Frag { uint4 q[2]; v16bf v; };

#define AS1 __attribute__((address_space(1)))
#define AS3 __attribute__((address_space(3)))

#if __has_builtin(__builtin_amdgcn_global_load_async_to_lds_b128) && \
    __has_builtin(__builtin_amdgcn_s_wait_asynccnt)
#define USE_ASYNC_LDS 1
#else
#define USE_ASYNC_LDS 0
#endif

// ---------------- helpers ----------------
__device__ __forceinline__ unsigned short bf16_rn(float f) {
  unsigned u = __float_as_uint(f);
  unsigned r = u + 0x7FFFu + ((u >> 16) & 1u);   // round-to-nearest-even
  return (unsigned short)(r >> 16);
}
__device__ __forceinline__ float bf16_f(unsigned short h) {
  return __uint_as_float(((unsigned)h) << 16);
}

// branchless insert into ascending 5-element list (keeps 5 smallest)
__device__ __forceinline__ void insert5(float t[5], float x) {
  float n4 = fminf(t[4], fmaxf(t[3], x));
  float n3 = fminf(t[3], fmaxf(t[2], x));
  float n2 = fminf(t[2], fmaxf(t[1], x));
  float n1 = fminf(t[1], fmaxf(t[0], x));
  float n0 = fminf(t[0], x);
  t[0] = n0; t[1] = n1; t[2] = n2; t[3] = n3; t[4] = n4;
}

__device__ __forceinline__ float gumbelf(unsigned i) {
  unsigned z = i * 0x9E3779B9u + 0xDEADBEEFu;   // deterministic hash "PRNG"
  z ^= z >> 16; z *= 0x7FEB352Du;
  z ^= z >> 15; z *= 0x846CA68Bu;
  z ^= z >> 16;
  float u = ((float)(z >> 8) + 0.5f) * (1.0f / 16777216.0f);  // (0,1)
  return -logf(-logf(u));
}

// ---------------- kernel 1: f32 -> (hi,lo) bf16 split + row sq-norms ----------------
__global__ void __launch_bounds__(256)
prep_kernel(const float* __restrict__ x, unsigned short* __restrict__ hi,
            unsigned short* __restrict__ lo, float* __restrict__ sq) {
  int row  = blockIdx.x * 8 + (threadIdx.x >> 5);  // one wave per 256-dim row
  int lane = threadIdx.x & 31;
  const float* xr = x + (size_t)row * kD + lane * 8;
  float4 a = ((const float4*)xr)[0];
  float4 c = ((const float4*)xr)[1];
  float v[8] = {a.x, a.y, a.z, a.w, c.x, c.y, c.z, c.w};
  unsigned hpk[4], lpk[4];
  float s = 0.f;
#pragma unroll
  for (int i = 0; i < 4; ++i) {
    float v0 = v[2 * i], v1 = v[2 * i + 1];
    unsigned short h0 = bf16_rn(v0), h1 = bf16_rn(v1);
    unsigned short l0 = bf16_rn(v0 - bf16_f(h0)), l1 = bf16_rn(v1 - bf16_f(h1));
    hpk[i] = (unsigned)h0 | ((unsigned)h1 << 16);
    lpk[i] = (unsigned)l0 | ((unsigned)l1 << 16);
    s += v0 * v0 + v1 * v1;
  }
  size_t off = (size_t)row * kD + lane * 8;
  ((uint4*)(hi + off))[0] = make_uint4(hpk[0], hpk[1], hpk[2], hpk[3]);
  ((uint4*)(lo + off))[0] = make_uint4(lpk[0], lpk[1], lpk[2], lpk[3]);
#pragma unroll
  for (int o = 16; o; o >>= 1) s += __shfl_xor(s, o, 32);
  if (lane == 0) sq[row] = s;
}

// ---------------- per-tile WMMA + top-5 update ----------------
__device__ __forceinline__ void tile_compute(
    const unsigned short* __restrict__ th, const unsigned short* __restrict__ tl,
    const float* __restrict__ cs, const Frag* aH, const Frag* aL,
    const float* rsq, float (*t)[5], int ct, int myRow0, int laneMod, int laneHi) {
  v8f acc = {};
#pragma unroll
  for (int kc = 0; kc < 8; ++kc) {
    // B 32x16 bf16 layout: lane<16 -> K {0..7,16..23}, lane>=16 -> K {8..15,24..31}, col = lane%16
    Frag bH, bL;
    int off = laneMod * kLdsStride + kc * 32 + laneHi * 8;
    bH.q[0] = ((const uint4*)&th[off])[0];
    bH.q[1] = ((const uint4*)&th[off + 16])[0];
    bL.q[0] = ((const uint4*)&tl[off])[0];
    bL.q[1] = ((const uint4*)&tl[off + 16])[0];
    // bf16x3 split: hi*hi + hi*lo + lo*hi ~ f32-accurate dot product
    acc = __builtin_amdgcn_wmma_f32_16x16x32_bf16(false, aH[kc].v, false, bH.v,
                                                  (short)0, acc, false, false);
    acc = __builtin_amdgcn_wmma_f32_16x16x32_bf16(false, aH[kc].v, false, bL.v,
                                                  (short)0, acc, false, false);
    acc = __builtin_amdgcn_wmma_f32_16x16x32_bf16(false, aL[kc].v, false, bH.v,
                                                  (short)0, acc, false, false);
  }
  int colG  = ct * 16 + laneMod;
  float csq = cs[laneMod];
#pragma unroll
  for (int v = 0; v < 8; ++v) {
    float d2 = fmaxf(rsq[v] + csq - 2.0f * acc[v], 0.0f);
    int rowG = myRow0 + v + 8 * laneHi;
    d2 = (rowG == colG) ? __builtin_inff() : d2;  // exclude self
    insert5(t[v], d2);
  }
}

// ---------------- kernel 2: fused WMMA Gram + 5th-NN distance ----------------
__global__ void __launch_bounds__(256)
knn_kernel(const unsigned short* __restrict__ hi, const unsigned short* __restrict__ lo,
           const float* __restrict__ sq, float* __restrict__ w) {
  __shared__ unsigned short tHi[2][16 * kLdsStride];
  __shared__ unsigned short tLo[2][16 * kLdsStride];
  __shared__ float colSq[2][16];

  const int b       = blockIdx.x >> 5;          // kN/kRowsPerBlock == 32
  const int rb      = blockIdx.x & 31;
  const int rowBase = rb * kRowsPerBlock;
  const int lane    = threadIdx.x & 31;
  const int wave    = threadIdx.x >> 5;
  const int laneMod = lane & 15;
  const int laneHi  = lane >> 4;                // which 16-lane half
  const int myRow0  = rowBase + wave * 16;      // 16 rows per wave
  const size_t batchOff = (size_t)b * kN * kD;

  // Preload this wave's A fragments (hi & lo) for all 8 K-chunks of D=256.
  // A 16x32 bf16 layout: lane<16 -> K {0..7,16..23}; lane>=16 -> K {8..15,24..31}.
  Frag aH[8], aL[8];
  {
    const unsigned short* rH = hi + batchOff + (size_t)(myRow0 + laneMod) * kD;
    const unsigned short* rL = lo + batchOff + (size_t)(myRow0 + laneMod) * kD;
#pragma unroll
    for (int kc = 0; kc < 8; ++kc) {
      int k0 = kc * 32 + laneHi * 8;
      aH[kc].q[0] = ((const uint4*)(rH + k0))[0];
      aH[kc].q[1] = ((const uint4*)(rH + k0 + 16))[0];
      aL[kc].q[0] = ((const uint4*)(rL + k0))[0];
      aL[kc].q[1] = ((const uint4*)(rL + k0 + 16))[0];
    }
  }
  float rsq[8];
#pragma unroll
  for (int v = 0; v < 8; ++v) rsq[v] = sq[b * kN + myRow0 + v + 8 * laneHi];

  float t[8][5];
#pragma unroll
  for (int v = 0; v < 8; ++v)
#pragma unroll
    for (int j = 0; j < 5; ++j) t[v][j] = __builtin_inff();

  const int sr   = threadIdx.x >> 4;          // staging row (0..15)
  const int scol = (threadIdx.x & 15) * 16;   // staging col offset (bf16 elems)

#if USE_ASYNC_LDS
  // ----- double-buffered, async global->LDS pipeline (latency hidden) -----
#define STAGE_TILE(ct_) do {                                                       \
    int buf_ = (ct_) & 1;                                                          \
    const unsigned short* gH_ = hi + batchOff + (size_t)((ct_) * 16 + sr) * kD + scol; \
    const unsigned short* gL_ = lo + batchOff + (size_t)((ct_) * 16 + sr) * kD + scol; \
    unsigned short* lH_ = &tHi[buf_][sr * kLdsStride + scol];                      \
    unsigned short* lL_ = &tLo[buf_][sr * kLdsStride + scol];                      \
    __builtin_amdgcn_global_load_async_to_lds_b128((AS1 v4i*)gH_,       (AS3 v4i*)lH_,       0, 0); \
    __builtin_amdgcn_global_load_async_to_lds_b128((AS1 v4i*)(gH_ + 8), (AS3 v4i*)(lH_ + 8), 0, 0); \
    __builtin_amdgcn_global_load_async_to_lds_b128((AS1 v4i*)gL_,       (AS3 v4i*)lL_,       0, 0); \
    __builtin_amdgcn_global_load_async_to_lds_b128((AS1 v4i*)(gL_ + 8), (AS3 v4i*)(lL_ + 8), 0, 0); \
  } while (0)

  STAGE_TILE(0);
  for (int ct = 0; ct < kNumTiles - 1; ++ct) {
    const int cur = ct & 1;
    STAGE_TILE(ct + 1);   // 4 async b128 in flight for next tile
    if (threadIdx.x < 16) colSq[cur][threadIdx.x] = sq[b * kN + ct * 16 + threadIdx.x];
    __builtin_amdgcn_s_wait_asynccnt(4);   // in-order: oldest 4 (tile ct) have landed
    __syncthreads();                       // all waves see tile ct in LDS
    tile_compute(tHi[cur], tLo[cur], colSq[cur], aH, aL, rsq, t,
                 ct, myRow0, laneMod, laneHi);
    __syncthreads();                       // tile ct buffer free for reuse
  }
  {
    const int ct = kNumTiles - 1, cur = ct & 1;
    if (threadIdx.x < 16) colSq[cur][threadIdx.x] = sq[b * kN + ct * 16 + threadIdx.x];
    __builtin_amdgcn_s_wait_asynccnt(0);
    __syncthreads();
    tile_compute(tHi[cur], tLo[cur], colSq[cur], aH, aL, rsq, t,
                 ct, myRow0, laneMod, laneHi);
  }
#undef STAGE_TILE
#else
  // ----- fallback: synchronous VGPR round-trip staging -----
  for (int ct = 0; ct < kNumTiles; ++ct) {
    __syncthreads();
    const unsigned short* gH = hi + batchOff + (size_t)(ct * 16 + sr) * kD + scol;
    const unsigned short* gL = lo + batchOff + (size_t)(ct * 16 + sr) * kD + scol;
    uint4* dH = (uint4*)&tHi[0][sr * kLdsStride + scol];
    uint4* dL = (uint4*)&tLo[0][sr * kLdsStride + scol];
    dH[0] = ((const uint4*)gH)[0]; dH[1] = ((const uint4*)gH)[1];
    dL[0] = ((const uint4*)gL)[0]; dL[1] = ((const uint4*)gL)[1];
    if (threadIdx.x < 16) colSq[0][threadIdx.x] = sq[b * kN + ct * 16 + threadIdx.x];
    if (ct + 1 < kNumTiles) {
      __builtin_prefetch(gH + 16 * kD, 0, 1);
      __builtin_prefetch(gL + 16 * kD, 0, 1);
    }
    __syncthreads();
    tile_compute(tHi[0], tLo[0], colSq[0], aH, aL, rsq, t,
                 ct, myRow0, laneMod, laneHi);
  }
#endif

  // merge partial top-5 lists across the 16 column-lanes of each half
#pragma unroll
  for (int v = 0; v < 8; ++v) {
#pragma unroll
    for (int o = 1; o < 16; o <<= 1) {
      float o0 = __shfl_xor(t[v][0], o, 32), o1 = __shfl_xor(t[v][1], o, 32),
            o2 = __shfl_xor(t[v][2], o, 32), o3 = __shfl_xor(t[v][3], o, 32),
            o4 = __shfl_xor(t[v][4], o, 32);
      insert5(t[v], o0); insert5(t[v], o1); insert5(t[v], o2);
      insert5(t[v], o3); insert5(t[v], o4);
    }
    if (laneMod == 0)
      w[b * kN + myRow0 + v + 8 * laneHi] = sqrtf(t[v][4] + 1e-12f);
  }
}

// ---------------- kernel 3: Gumbel top-256 per batch ----------------
__global__ void __launch_bounds__(256)
topk_kernel(const float* __restrict__ w, int* __restrict__ idxOut) {
  __shared__ float scb[kN];
  __shared__ float rv[256];
  __shared__ int   ri[256];
  int b = blockIdx.x, tid = threadIdx.x;
  for (int i = tid; i < kN; i += 256)
    scb[i] = logf(w[b * kN + i] + 1e-12f) + gumbelf((unsigned)(b * kN + i));
  __syncthreads();
  for (int s = 0; s < kSampleK; ++s) {
    float best = -__builtin_inff(); int bi = kN;
    for (int i = tid; i < kN; i += 256) {
      float v = scb[i];
      if (v > best) { best = v; bi = i; }
    }
    rv[tid] = best; ri[tid] = bi;
    __syncthreads();
    for (int o = 128; o; o >>= 1) {
      if (tid < o) {
        float v2 = rv[tid + o]; int i2 = ri[tid + o];
        if (v2 > rv[tid] || (v2 == rv[tid] && i2 < ri[tid])) { rv[tid] = v2; ri[tid] = i2; }
      }
      __syncthreads();
    }
    if (tid == 0) { idxOut[b * kSampleK + s] = ri[0]; scb[ri[0]] = -__builtin_inff(); }
    __syncthreads();
  }
}

// ---------------- kernel 4: gather sampled tokens + scalar tail ----------------
__global__ void __launch_bounds__(64)
gather_kernel(const float* __restrict__ x, const int* __restrict__ idx,
              float* __restrict__ out) {
  int s = blockIdx.x;                 // 0 .. B*SAMPLE_K-1
  int b = s >> 8;                     // kSampleK == 256
  int j = idx[s];
  const float4* src = (const float4*)(x + ((size_t)b * kN + j) * kD);
  float4* dst = (float4*)(out + (size_t)s * kD);
  dst[threadIdx.x] = src[threadIdx.x];           // 64 * float4 = 256 floats
  if (s == 0 && threadIdx.x == 0)
    out[(size_t)kB * kSampleK * kD] = 0.0f;      // second tuple element
}

// ---------------- launch ----------------
extern "C" void kernel_launch(void* const* d_in, const int* in_sizes, int n_in,
                              void* d_out, int out_size, void* d_ws, size_t ws_size,
                              hipStream_t stream) {
  (void)in_sizes; (void)n_in; (void)out_size; (void)ws_size;
  const float* x = (const float*)d_in[0];
  float* out = (float*)d_out;

  char* ws = (char*)d_ws;
  size_t o = 0;
  unsigned short* hi = (unsigned short*)(ws + o); o += (size_t)kB * kN * kD * 2;
  unsigned short* lo = (unsigned short*)(ws + o); o += (size_t)kB * kN * kD * 2;
  float* sq  = (float*)(ws + o); o += (size_t)kB * kN * 4;
  float* wgt = (float*)(ws + o); o += (size_t)kB * kN * 4;
  int*   idx = (int*)(ws + o);

  prep_kernel<<<kB * kN / 8, 256, 0, stream>>>(x, hi, lo, sq);
  knn_kernel<<<kB * (kN / kRowsPerBlock), 256, 0, stream>>>(hi, lo, sq, wgt);
  topk_kernel<<<kB, 256, 0, stream>>>(wgt, idx);
  gather_kernel<<<kB * kSampleK, 64, 0, stream>>>(x, idx, out);
}
